// CrossAttention_3246995275787
// MI455X (gfx1250) — matrix operand order repaired and checked by reading, not compiled, not run on hardware
//
#include <hip/hip_runtime.h>
#include <hip/hip_bf16.h>
#include <stdint.h>

typedef __bf16 bh;
typedef __attribute__((ext_vector_type(16))) __bf16 v16bf;
typedef __attribute__((ext_vector_type(8)))  float  v8f;

static constexpr int kDim     = 1024;
static constexpr int kHeads   = 16;
static constexpr int kDimHead = 64;
static constexpr int kB       = 4;
static constexpr int kN       = 1024;
static constexpr int kJ       = 1024;
static constexpr int kFF      = 4096;
static constexpr int kRows    = kB * kN;   // 4096 flattened rows

// -------------------------------------------------------------------------
// LayerNorm over last dim (D=1024) + cast to bf16. One 256-thread block/row.
// -------------------------------------------------------------------------
__global__ __launch_bounds__(256)
void ln_cast_kernel(const float* __restrict__ x, const float* __restrict__ g,
                    const float* __restrict__ b, bh* __restrict__ out, int D) {
  const int row = blockIdx.x;
  const float* xr = x + (size_t)row * D;
  float s = 0.f, s2 = 0.f;
  for (int i = threadIdx.x; i < D; i += 256) { float v = xr[i]; s += v; s2 += v * v; }
  __shared__ float rs[256], rq[256];
  rs[threadIdx.x] = s; rq[threadIdx.x] = s2;
  __syncthreads();
  for (int off = 128; off > 0; off >>= 1) {
    if (threadIdx.x < off) { rs[threadIdx.x] += rs[threadIdx.x + off];
                             rq[threadIdx.x] += rq[threadIdx.x + off]; }
    __syncthreads();
  }
  const float mu  = rs[0] / D;
  const float var = rq[0] / D - mu * mu;
  const float inv = rsqrtf(var + 1e-5f);
  bh* orow = out + (size_t)row * D;
  for (int i = threadIdx.x; i < D; i += 256)
    orow[i] = (bh)((xr[i] - mu) * inv * g[i] + b[i]);
}

// -------------------------------------------------------------------------
// Cast fp32 weight [K x N] -> bf16 transposed [N x K] so WMMA B-fragment
// loads (column n, contiguous k) are dword loads.
// -------------------------------------------------------------------------
__global__ __launch_bounds__(256)
void cast_transpose_kernel(const float* __restrict__ W, bh* __restrict__ Wt,
                           int K, int N) {
  size_t idx = (size_t)blockIdx.x * 256 + threadIdx.x;
  if (idx >= (size_t)K * N) return;
  const int k = (int)(idx / N), n = (int)(idx % N);
  Wt[(size_t)n * K + k] = (bh)W[idx];
}

// -------------------------------------------------------------------------
// WMMA GEMM: C[M x N] = alpha * A[M x K](bf16) @ Bt[N x K](bf16)^T
// 128 threads = 4 waves; each wave computes a 32x64 C tile:
//   2 A fragments x 4 B fragments -> 8 WMMAs per 32-wide K step,
//   every B fragment reused across both row tiles.
// SWIGLU: also computes gate tile at Bt row offset gate_off, emits
//   silu(gate) * val. OUT_BF16 / ACCUM select epilogue.
// -------------------------------------------------------------------------
template <bool SWIGLU, bool OUT_BF16, bool ACCUM>
__global__ __launch_bounds__(128)
void wmma_gemm_kernel(const bh* __restrict__ A, const bh* __restrict__ Bt,
                      void* __restrict__ Cout, int M, int N, int K,
                      float alpha, int gate_off) {
  const int lane = threadIdx.x & 31;
  const int wave = threadIdx.x >> 5;
  const int m16  = lane & 15;
  const int half = lane >> 4;
  const int row0 = blockIdx.y * 128 + wave * 32;
  const int col0 = blockIdx.x * 64;
  (void)M;

  v8f acc[2][4]  = {};
  v8f accg[2][4] = {};

  const bh* arow[2] = { A + (size_t)(row0 + m16) * K,
                        A + (size_t)(row0 + 16 + m16) * K };
  for (int k0 = 0; k0 < K; k0 += 32) {
    // A fragments: contiguous k order (same as B), kb = half*16 + 2*v
    v16bf af[2];
#pragma unroll
    for (int rt = 0; rt < 2; ++rt) {
      uint32_t* a32 = reinterpret_cast<uint32_t*>(&af[rt]);
#pragma unroll
      for (int v = 0; v < 8; ++v) {
        const int kb = half * 16 + (v << 1);
        a32[v] = *reinterpret_cast<const uint32_t*>(arow[rt] + k0 + kb);
      }
      if (k0 + 256 < K) __builtin_prefetch(arow[rt] + k0 + 256, 0, 1);  // global_prefetch_b8
    }

#pragma unroll
    for (int t = 0; t < 4; ++t) {
      // B fragment: lane=n, k = half*16 + 2v (+1); contiguous in Bt row.
      v16bf bfrag;
      uint32_t* b32 = reinterpret_cast<uint32_t*>(&bfrag);
      const bh* brow = Bt + (size_t)(col0 + t * 16 + m16) * K + k0 + half * 16;
#pragma unroll
      for (int v = 0; v < 8; ++v)
        b32[v] = *reinterpret_cast<const uint32_t*>(brow + v * 2);
#pragma unroll
      for (int rt = 0; rt < 2; ++rt)
        acc[rt][t] = __builtin_amdgcn_wmma_f32_16x16x32_bf16(
            false, af[rt], false, bfrag, (short)0, acc[rt][t], false, false);
      if (SWIGLU) {
        v16bf gfrag;
        uint32_t* g32 = reinterpret_cast<uint32_t*>(&gfrag);
        const bh* grow = Bt + (size_t)(gate_off + col0 + t * 16 + m16) * K
                         + k0 + half * 16;
#pragma unroll
        for (int v = 0; v < 8; ++v)
          g32[v] = *reinterpret_cast<const uint32_t*>(grow + v * 2);
#pragma unroll
        for (int rt = 0; rt < 2; ++rt)
          accg[rt][t] = __builtin_amdgcn_wmma_f32_16x16x32_bf16(
              false, af[rt], false, gfrag, (short)0, accg[rt][t], false, false);
      }
    }
  }

  // Epilogue: C layout lane=n, VGPR r -> m = r + 8*half.
#pragma unroll
  for (int rt = 0; rt < 2; ++rt) {
#pragma unroll
    for (int t = 0; t < 4; ++t) {
#pragma unroll
      for (int r = 0; r < 8; ++r) {
        const int row = row0 + rt * 16 + r + half * 8;
        const int col = col0 + t * 16 + m16;
        float v = acc[rt][t][r] * alpha;
        if (SWIGLU) {
          const float gt = accg[rt][t][r];
          v = v * (gt / (1.f + __expf(-gt)));   // silu(gate) * val
        }
        const size_t idx = (size_t)row * N + col;
        if (OUT_BF16) {
          ((bh*)Cout)[idx] = (bh)v;
        } else {
          float* C = (float*)Cout;
          C[idx] = ACCUM ? (C[idx] + v) : v;
        }
      }
    }
  }
}

// -------------------------------------------------------------------------
// Multi-query flash attention. One wave per (b, h, 32-query tile); the two
// 16-row subtiles share every k/v B fragment (halves kv L2 traffic).
// q: [b*n, 1024] bf16 (pre-scaled), kv: [b*j, 128] bf16 (k=0..63, v=64..127),
// out: [b*n, 1024] bf16. Streams keys in tiles of 32 with online softmax.
// -------------------------------------------------------------------------
__global__ __launch_bounds__(128)
void mqa_attention_kernel(const bh* __restrict__ q, const bh* __restrict__ kv,
                          bh* __restrict__ o) {
  __shared__ bh psh[4][2][16][32];  // per-wave probability staging (C->A relayout)
  const int lane = threadIdx.x & 31;
  const int wave = threadIdx.x >> 5;
  const int m16  = lane & 15;
  const int half = lane >> 4;
  const int task = blockIdx.x * 4 + wave;   // 2048 tasks total
  const int mt = task & 31;                 // 32 query tiles of 32 rows
  const int h  = (task >> 5) & 15;
  const int b  = task >> 9;

  // q A fragments: [rowtile][kchunk], 16x64 -> two 16x32 fragments each
  v16bf qa[2][2];
#pragma unroll
  for (int rt = 0; rt < 2; ++rt) {
    const bh* qrow = q + (size_t)(b * kN + mt * 32 + rt * 16 + m16)
                         * (kHeads * kDimHead) + h * kDimHead;
#pragma unroll
    for (int kc = 0; kc < 2; ++kc) {
      uint32_t* a32 = reinterpret_cast<uint32_t*>(&qa[rt][kc]);
#pragma unroll
      for (int v = 0; v < 8; ++v) {
        const int kb = kc * 32 + half * 16 + (v << 1);
        a32[v] = *reinterpret_cast<const uint32_t*>(qrow + kb);
      }
    }
  }

  const bh* kvb = kv + (size_t)b * kJ * 128;
  v8f o4[2][4] = {};
  float rmax[2][8], rsum[2][8];
#pragma unroll
  for (int rt = 0; rt < 2; ++rt)
#pragma unroll
    for (int r = 0; r < 8; ++r) { rmax[rt][r] = -1e30f; rsum[rt][r] = 0.f; }

  for (int j0 = 0; j0 < kJ; j0 += 32) {
    // sim: 32 queries x 32 keys = [rowtile][keytile] fragments, K=64
    v8f s[2][2] = {};
#pragma unroll
    for (int kc = 0; kc < 2; ++kc) {
      v16bf b0, b1;
      uint32_t* p0 = reinterpret_cast<uint32_t*>(&b0);
      uint32_t* p1 = reinterpret_cast<uint32_t*>(&b1);
      const bh* kr0 = kvb + (size_t)(j0 + m16) * 128 + kc * 32 + half * 16;
      const bh* kr1 = kvb + (size_t)(j0 + 16 + m16) * 128 + kc * 32 + half * 16;
#pragma unroll
      for (int v = 0; v < 8; ++v) {
        p0[v] = *reinterpret_cast<const uint32_t*>(kr0 + v * 2);
        p1[v] = *reinterpret_cast<const uint32_t*>(kr1 + v * 2);
      }
#pragma unroll
      for (int rt = 0; rt < 2; ++rt) {
        s[rt][0] = __builtin_amdgcn_wmma_f32_16x16x32_bf16(
            false, qa[rt][kc], false, b0, (short)0, s[rt][0], false, false);
        s[rt][1] = __builtin_amdgcn_wmma_f32_16x16x32_bf16(
            false, qa[rt][kc], false, b1, (short)0, s[rt][1], false, false);
      }
    }
    // online softmax; row m = r + 8*half lives across the 16 lanes of a half
#pragma unroll
    for (int rt = 0; rt < 2; ++rt) {
#pragma unroll
      for (int r = 0; r < 8; ++r) {
        float tm = fmaxf(s[rt][0][r], s[rt][1][r]);
#pragma unroll
        for (int msk = 8; msk >= 1; msk >>= 1)
          tm = fmaxf(tm, __shfl_xor(tm, msk, 32));
        const float nm   = fmaxf(rmax[rt][r], tm);
        const float corr = __expf(rmax[rt][r] - nm);
        rmax[rt][r] = nm;
        const float e0 = __expf(s[rt][0][r] - nm);
        const float e1 = __expf(s[rt][1][r] - nm);
        float rrs = e0 + e1;
#pragma unroll
        for (int msk = 8; msk >= 1; msk >>= 1)
          rrs += __shfl_xor(rrs, msk, 32);
        rsum[rt][r] = rsum[rt][r] * corr + rrs;
#pragma unroll
        for (int t = 0; t < 4; ++t) o4[rt][t][r] *= corr;
        const int row = r + half * 8;
        psh[wave][rt][row][m16]      = (bh)e0;
        psh[wave][rt][row][16 + m16] = (bh)e1;
      }
    }
    __syncthreads();
    // re-load probabilities in A layout
    v16bf pa[2];
#pragma unroll
    for (int rt = 0; rt < 2; ++rt) {
      uint32_t* pp = reinterpret_cast<uint32_t*>(&pa[rt]);
#pragma unroll
      for (int v = 0; v < 8; ++v) {
        const int kb = half * 16 + (v << 1);
        pp[v] = *reinterpret_cast<const uint32_t*>(&psh[wave][rt][m16][kb]);
      }
    }
    // o += p @ v  (v tile 32x64, gathered transposed for B layout; shared
    // across both row tiles)
#pragma unroll
    for (int t = 0; t < 4; ++t) {
      v16bf vb;
#pragma unroll
      for (int v = 0; v < 8; ++v) {
#pragma unroll
        for (int e = 0; e < 2; ++e) {
          const int kk = half * 16 + v * 2 + e;
          vb[v * 2 + e] = kvb[(size_t)(j0 + kk) * 128 + 64 + t * 16 + m16];
        }
      }
#pragma unroll
      for (int rt = 0; rt < 2; ++rt)
        o4[rt][t] = __builtin_amdgcn_wmma_f32_16x16x32_bf16(
            false, pa[rt], false, vb, (short)0, o4[rt][t], false, false);
    }
    __syncthreads();
  }

  // normalize + store bf16, multi-head interleaved [b*n, h*64]
  bh* ob = o + (size_t)b * kN * (kHeads * kDimHead);
#pragma unroll
  for (int rt = 0; rt < 2; ++rt) {
#pragma unroll
    for (int r = 0; r < 8; ++r) {
      const float inv = 1.f / rsum[rt][r];
      const int row = mt * 32 + rt * 16 + r + half * 8;
#pragma unroll
      for (int t = 0; t < 4; ++t)
        ob[(size_t)row * 1024 + h * 64 + t * 16 + m16] = (bh)(o4[rt][t][r] * inv);
    }
  }
}

// -------------------------------------------------------------------------
extern "C" void kernel_launch(void* const* d_in, const int* in_sizes, int n_in,
                              void* d_out, int out_size, void* d_ws, size_t ws_size,
                              hipStream_t stream) {
  (void)in_sizes; (void)n_in; (void)out_size; (void)ws_size;
  const float* x    = (const float*)d_in[0];
  const float* ctx  = (const float*)d_in[1];
  const float* lnxg = (const float*)d_in[2];
  const float* lnxb = (const float*)d_in[3];
  const float* lncg = (const float*)d_in[4];
  const float* lncb = (const float*)d_in[5];
  const float* Wq   = (const float*)d_in[6];
  const float* Wkv  = (const float*)d_in[7];
  const float* Wout = (const float*)d_in[8];
  const float* Wff1 = (const float*)d_in[9];
  const float* Wff2 = (const float*)d_in[10];
  float* out = (float*)d_out;

  char* wp = (char*)d_ws;
  auto alloc = [&](size_t bytes) {
    char* p = wp; wp += (bytes + 255) & ~(size_t)255; return p;
  };
  bh* xn    = (bh*)alloc((size_t)kRows * kDim * 2);          // 8 MiB
  bh* cn    = (bh*)alloc((size_t)kRows * kDim * 2);          // 8 MiB
  bh* WqT   = (bh*)alloc((size_t)kDim * kDim * 2);           // 2 MiB
  bh* WkvT  = (bh*)alloc((size_t)128 * kDim * 2);            // 0.25 MiB
  bh* WoutT = (bh*)alloc((size_t)kDim * kDim * 2);           // 2 MiB
  bh* Wff1T = (bh*)alloc((size_t)(2 * kFF) * kDim * 2);      // 16 MiB
  bh* Wff2T = (bh*)alloc((size_t)kDim * kFF * 2);            // 8 MiB
  bh* qb    = (bh*)alloc((size_t)kRows * kDim * 2);          // 8 MiB
  bh* kvb   = (bh*)alloc((size_t)kRows * 128 * 2);           // 1 MiB
  bh* ao    = (bh*)alloc((size_t)kRows * kDim * 2);          // 8 MiB
  bh* hb    = (bh*)alloc((size_t)kRows * kFF * 2);           // 32 MiB

  // 1) LayerNorms -> bf16
  ln_cast_kernel<<<kRows, 256, 0, stream>>>(x,   lnxg, lnxb, xn, kDim);
  ln_cast_kernel<<<kRows, 256, 0, stream>>>(ctx, lncg, lncb, cn, kDim);

  // 2) Weight cast+transpose to [N x K] bf16
  cast_transpose_kernel<<<(kDim * kDim + 255) / 256, 256, 0, stream>>>(Wq,   WqT,   kDim, kDim);
  cast_transpose_kernel<<<(kDim * 128  + 255) / 256, 256, 0, stream>>>(Wkv,  WkvT,  kDim, 128);
  cast_transpose_kernel<<<(kDim * kDim + 255) / 256, 256, 0, stream>>>(Wout, WoutT, kDim, kDim);
  cast_transpose_kernel<<<(kDim * 2 * kFF + 255) / 256, 256, 0, stream>>>(Wff1, Wff1T, kDim, 2 * kFF);
  cast_transpose_kernel<<<(kFF * kDim + 255) / 256, 256, 0, stream>>>(Wff2, Wff2T, kFF, kDim);

  // 3) q = xn @ Wq, fused 1/sqrt(d) scale, bf16 out
  wmma_gemm_kernel<false, true, false><<<dim3(kDim / 64, kRows / 128), 128, 0, stream>>>(
      xn, WqT, qb, kRows, kDim, kDim, 0.125f, 0);
  // 4) kv = cn @ Wkv
  wmma_gemm_kernel<false, true, false><<<dim3(128 / 64, kRows / 128), 128, 0, stream>>>(
      cn, WkvT, kvb, kRows, 128, kDim, 1.f, 0);
  // 5) flash MQA attention (2048 wave-tasks, 4 waves/block)
  mqa_attention_kernel<<<(kB * kHeads * (kN / 32)) / 4, 128, 0, stream>>>(qb, kvb, ao);
  // 6) h = silu(gate) * val, fused SwiGLU GEMM (gate at Bt row offset 4096)
  wmma_gemm_kernel<true, true, false><<<dim3(kFF / 64, kRows / 128), 128, 0, stream>>>(
      xn, Wff1T, hb, kRows, kFF, kDim, 1.f, kFF);
  // 7) out = attn_out @ Wout (f32 write)
  wmma_gemm_kernel<false, false, false><<<dim3(kDim / 64, kRows / 128), 128, 0, stream>>>(
      ao, WoutT, out, kRows, kDim, kDim, 1.f, 0);
  // 8) out += h @ Wff2 (f32 accumulate)
  wmma_gemm_kernel<false, false, true><<<dim3(kDim / 64, kRows / 128), 128, 0, stream>>>(
      hb, Wff2T, out, kRows, kDim, kFF, 1.f, 0);
}
